// ImgNetHY_55224689492718
// MI455X (gfx1250) — matrix operand order repaired and checked by reading, not compiled, not run on hardware
//
#include <hip/hip_runtime.h>
#include <hip/hip_bf16.h>

// ---------------------------------------------------------------------------
// ImgNetHY hypergraph conv pipeline for MI455X (gfx1250, wave32, WMMA).
//   1. row-normalize x -> xn (bf16) + norms
//   2. transpose W1 -> W1t (bf16), W2 -> W2t (bf16)
//   3. fused S = xn*xn^T GEMM (v_wmma_f32_16x16x32_bf16, 4 n-tiles/A-frag)
//      + per-row top-8 (S never hits HBM)
//   4. deg via float atomics
//   5. hW = diag(norm) * (xn @ W1)   (WMMA, 2x4 register blocking)
//   6. edge-mean gather, atomic scatter, /deg + b1, ReLU -> feat (bf16)
//   7. hW2 = feat @ W2 (WMMA), edge-mean, scatter, tanh(/deg + b2) -> out
// ---------------------------------------------------------------------------

typedef __attribute__((ext_vector_type(16))) __bf16 v16bf;
typedef __attribute__((ext_vector_type(8)))  float  v8f;

static constexpr int kN = 4096;   // nodes / hyperedges
static constexpr int kF = 4096;   // input feature dim
static constexpr int kB = 4096;   // hidden dim
static constexpr int kC = 64;     // code length
static constexpr int kK = 8;      // neighbors per hyperedge

union FragU { v16bf v; uint4 q[2]; };

// A-matrix 16x32 bf16 fragment: lane L (row M = L&15) holds K = koff..koff+7
// and koff+16..koff+23 where koff = (L>=16)*8.  p points at row_base+k+koff.
__device__ inline v16bf load_fragA(const __hip_bfloat16* p) {
  FragU u;
  u.q[0] = *(const uint4*)(p);
  u.q[1] = *(const uint4*)(p + 16);
  return u.v;
}

// B-matrix 32x16 bf16 fragment: lane L (col N = L&15) holds K = koff..koff+7
// and koff+16..koff+23 where koff = (L>=16)*8.  p points at col_base + k + koff_b.
__device__ inline v16bf load_fragB(const __hip_bfloat16* p) {
  FragU u;
  u.q[0] = *(const uint4*)(p);
  u.q[1] = *(const uint4*)(p + 16);
  return u.v;
}

__device__ inline v8f wmma_bf16(v16bf a, v16bf b, v8f c) {
  return __builtin_amdgcn_wmma_f32_16x16x32_bf16(
      /*neg_a=*/false, a, /*neg_b=*/false, b,
      /*c_mod=*/(short)0, c, /*reuse_a=*/false, /*reuse_b=*/false);
}

__device__ inline void topk_insert(float (&v)[kK], int (&ix)[kK], float nv, int ni) {
  if (nv <= v[kK - 1]) return;
  v[kK - 1] = nv; ix[kK - 1] = ni;
#pragma unroll
  for (int t = kK - 1; t > 0; --t) {
    if (v[t] > v[t - 1]) {
      float tv = v[t]; v[t] = v[t - 1]; v[t - 1] = tv;
      int   ti = ix[t]; ix[t] = ix[t - 1]; ix[t - 1] = ti;
    }
  }
}

// ------------------------- kernel 1: normalize ------------------------------
__global__ void hy_normalize_kernel(const float* __restrict__ x,
                                    __hip_bfloat16* __restrict__ xn,
                                    float* __restrict__ norms) {
  __shared__ float red[256];
  const int row = blockIdx.x;
  const float* xr = x + (size_t)row * kF;
  float s = 0.f;
  for (int c = threadIdx.x; c < kF; c += blockDim.x) { float v = xr[c]; s += v * v; }
  red[threadIdx.x] = s;
  __syncthreads();
  for (int off = 128; off > 0; off >>= 1) {
    if ((int)threadIdx.x < off) red[threadIdx.x] += red[threadIdx.x + off];
    __syncthreads();
  }
  float nrm = fmaxf(sqrtf(red[0]), 1e-12f);
  if (threadIdx.x == 0) norms[row] = nrm;
  const float inv = 1.0f / nrm;
  for (int c = threadIdx.x; c < kF; c += blockDim.x)
    xn[(size_t)row * kF + c] = __float2bfloat16(xr[c] * inv);
}

// ------------------- kernel 2: transpose fp32 -> bf16 -----------------------
// W: rows x cols (row-major fp32)  ->  Wt: cols x rows (row-major bf16)
__global__ void hy_transpose_bf16_kernel(const float* __restrict__ W,
                                         __hip_bfloat16* __restrict__ Wt,
                                         int rows, int cols) {
  size_t id = (size_t)blockIdx.x * blockDim.x + threadIdx.x;
  size_t total = (size_t)rows * cols;
  if (id >= total) return;
  size_t k = id / cols;
  size_t n = id % cols;
  Wt[n * (size_t)rows + k] = __float2bfloat16(W[id]);
}

// --------------- kernel 3: fused S = xn*xn^T + per-row top-8 ----------------
// grid: 256 blocks (16-row slabs), 256 threads = 8 waves.
// Each wave owns 64-column groups jg = wave, wave+8, ... (8 groups of 4
// 16-col tiles); one A-frag load feeds 4 WMMAs.
__global__ void hy_s_topk_kernel(const __hip_bfloat16* __restrict__ xn,
                                 int* __restrict__ topk) {
  __shared__ float tile[8][16][68];      // 4 jtiles wide, +4 pad for banks
  __shared__ float mergeV[8][16][kK];
  __shared__ int   mergeI[8][16][kK];

  const int lane  = threadIdx.x & 31;
  const int wave  = threadIdx.x >> 5;
  const int m0    = blockIdx.x * 16;
  const int m     = lane & 15;
  const int koffA = (lane >> 4) * 8;
  const int koffB = (lane >> 4) * 8;

  float bestV[kK]; int bestI[kK];
#pragma unroll
  for (int t = 0; t < kK; ++t) { bestV[t] = -3.0e38f; bestI[t] = 0; }

  const __hip_bfloat16* aRow = xn + (size_t)(m0 + m) * kF + koffA;
  const int jgroups = kN / 64;   // 64 groups of 4 tiles

  for (int jg = wave; jg < jgroups; jg += 8) {
    const int n0 = jg * 64;
    const __hip_bfloat16* bRow = xn + (size_t)(n0 + m) * kF + koffB;
    v8f acc[4] = {};
    for (int k = 0; k < kF; k += 32) {
      v16bf af = load_fragA(aRow + k);
#pragma unroll
      for (int j = 0; j < 4; ++j) {
        v16bf bf = load_fragB(bRow + (size_t)j * 16 * kF + k);
        acc[j] = wmma_bf16(af, bf, acc[j]);
      }
    }
    // C frags -> LDS: lane holds col N = lane&15, rows M = r + 8*(lane>=16)
    const int colN  = lane & 15;
    const int rbase = (lane >> 4) * 8;
#pragma unroll
    for (int j = 0; j < 4; ++j)
#pragma unroll
      for (int r = 0; r < 8; ++r) tile[wave][rbase + r][j * 16 + colN] = acc[j][r];
    __syncthreads();
    if (lane < 16) {
      for (int jj = 0; jj < 64; ++jj)
        topk_insert(bestV, bestI, tile[wave][lane][jj], n0 + jj);
    }
    __syncthreads();
  }

  if (lane < 16) {
#pragma unroll
    for (int t = 0; t < kK; ++t) { mergeV[wave][lane][t] = bestV[t]; mergeI[wave][lane][t] = bestI[t]; }
  }
  __syncthreads();

  if (threadIdx.x < 16) {
    const int row = threadIdx.x;
    float fv[kK]; int fi[kK];
#pragma unroll
    for (int t = 0; t < kK; ++t) { fv[t] = -3.0e38f; fi[t] = 0; }
    for (int w = 0; w < 8; ++w)
      for (int t = 0; t < kK; ++t)
        topk_insert(fv, fi, mergeV[w][row][t], mergeI[w][row][t]);
#pragma unroll
    for (int t = 0; t < kK; ++t) topk[(m0 + row) * kK + t] = fi[t];
  }
}

// --------------------------- kernel 4: zero ---------------------------------
__global__ void hy_zero_f32_kernel(float* __restrict__ p, size_t n) {
  size_t id = (size_t)blockIdx.x * blockDim.x + threadIdx.x;
  if (id < n) p[id] = 0.f;
}

// ----------------------- kernel 5: degree scatter ---------------------------
__global__ void hy_deg_kernel(const int* __restrict__ topk, float* __restrict__ deg) {
  int id = blockIdx.x * blockDim.x + threadIdx.x;
  if (id < kN * kK) atomicAdd(&deg[topk[id]], 1.0f);
}

// ------------------ kernel 6: generic bf16 WMMA GEMM ------------------------
// C[m][n] = scale(m) * sum_k A[m][k] * Bt[n][k]
// Register blocked: each wave computes a 32x64 block (2 m-tiles x 4 n-tiles,
// 8 accumulators); one A-frag pair feeds 8 WMMAs per 4 B-frags.
// grid.x = M/32 slabs; (blockIdx.y * waves + wave) = 64-col n-block index.
__global__ void hy_gemm_bf16_kernel(const __hip_bfloat16* __restrict__ A,
                                    const __hip_bfloat16* __restrict__ Bt,
                                    float* __restrict__ C, int ldc,
                                    int Kdim, int nBlocksTotal,
                                    const float* __restrict__ rowScale) {
  const int lane  = threadIdx.x & 31;
  const int wave  = threadIdx.x >> 5;
  const int waves = blockDim.x >> 5;
  const int nblk  = blockIdx.y * waves + wave;
  if (nblk >= nBlocksTotal) return;            // wave-uniform exit
  const int m0 = blockIdx.x * 32;
  const int n0 = nblk * 64;
  const int m  = lane & 15;
  const __hip_bfloat16* aRow0 = A + (size_t)(m0 + m) * Kdim + (lane >> 4) * 8;
  const __hip_bfloat16* aRow1 = aRow0 + (size_t)16 * Kdim;
  const __hip_bfloat16* bRow  = Bt + (size_t)(n0 + m) * Kdim + (lane >> 4) * 8;

  v8f acc[2][4] = {};
  for (int k = 0; k < Kdim; k += 32) {
    v16bf a0 = load_fragA(aRow0 + k);
    v16bf a1 = load_fragA(aRow1 + k);
#pragma unroll
    for (int j = 0; j < 4; ++j) {
      v16bf bf = load_fragB(bRow + (size_t)j * 16 * Kdim + k);
      acc[0][j] = wmma_bf16(a0, bf, acc[0][j]);
      acc[1][j] = wmma_bf16(a1, bf, acc[1][j]);
    }
  }

  const int colBase = n0 + (lane & 15);
  const int rOff    = (lane >> 4) << 3;
#pragma unroll
  for (int mi = 0; mi < 2; ++mi) {
    const int rbase = m0 + mi * 16 + rOff;
#pragma unroll
    for (int r = 0; r < 8; ++r) {
      const float scale = rowScale ? rowScale[rbase + r] : 1.0f;
#pragma unroll
      for (int j = 0; j < 4; ++j)
        C[(size_t)(rbase + r) * ldc + colBase + j * 16] = acc[mi][j][r] * scale;
    }
  }
}

// ------------------- kernel 7: hyperedge mean (gather) ----------------------
__global__ void hy_edge_mean_kernel(const float* __restrict__ hW,
                                    const int* __restrict__ topk,
                                    float* __restrict__ em, int cols) {
  const int e = blockIdx.x;
  int nk[kK];
#pragma unroll
  for (int k = 0; k < kK; ++k) nk[k] = topk[e * kK + k];
  for (int c = threadIdx.x; c < cols; c += blockDim.x) {
    float s = 0.f;
#pragma unroll
    for (int k = 0; k < kK; ++k) s += hW[(size_t)nk[k] * cols + c];
    em[(size_t)e * cols + c] = s * (1.0f / kK);
  }
}

// --------------------- kernel 8: node scatter (atomics) ---------------------
__global__ void hy_scatter_kernel(const float* __restrict__ em,
                                  const int* __restrict__ topk,
                                  float* __restrict__ acc, int cols) {
  const int e = blockIdx.x;
  int nk[kK];
#pragma unroll
  for (int k = 0; k < kK; ++k) nk[k] = topk[e * kK + k];
  for (int c = threadIdx.x; c < cols; c += blockDim.x) {
    float v = em[(size_t)e * cols + c];
#pragma unroll
    for (int k = 0; k < kK; ++k) atomicAdd(&acc[(size_t)nk[k] * cols + c], v);
  }
}

// --------------- kernel 9: /deg + bias + ReLU -> bf16 feat ------------------
__global__ void hy_finish1_kernel(const float* __restrict__ acc,
                                  const float* __restrict__ deg,
                                  const float* __restrict__ b1,
                                  __hip_bfloat16* __restrict__ feat) {
  const int i = blockIdx.x;
  const float invd = 1.0f / fmaxf(deg[i], 1.0f);
  for (int c = threadIdx.x; c < kB; c += blockDim.x) {
    float v = acc[(size_t)i * kB + c] * invd + b1[c];
    feat[(size_t)i * kB + c] = __float2bfloat16(fmaxf(v, 0.f));
  }
}

// ------------------ kernel 10: /deg + bias + tanh -> out --------------------
__global__ void hy_finish2_kernel(const float* __restrict__ acc,
                                  const float* __restrict__ deg,
                                  const float* __restrict__ b2,
                                  float* __restrict__ out) {
  const int i = blockIdx.x;
  const float invd = 1.0f / fmaxf(deg[i], 1.0f);
  for (int c = threadIdx.x; c < kC; c += blockDim.x) {
    float v = acc[(size_t)i * kC + c] * invd + b2[c];
    out[(size_t)i * kC + c] = tanhf(v);
  }
}

// ---------------------------------------------------------------------------
extern "C" void kernel_launch(void* const* d_in, const int* in_sizes, int n_in,
                              void* d_out, int out_size, void* d_ws, size_t ws_size,
                              hipStream_t stream) {
  (void)in_sizes; (void)n_in; (void)out_size; (void)ws_size;

  const float* x  = (const float*)d_in[0];
  const float* W1 = (const float*)d_in[1];
  const float* b1 = (const float*)d_in[2];
  const float* W2 = (const float*)d_in[3];
  const float* b2 = (const float*)d_in[4];
  float* out = (float*)d_out;

  char* base = (char*)d_ws;
  size_t off = 0;
  auto alloc = [&](size_t bytes) -> char* {
    char* p = base + off;
    off = (off + bytes + 255) & ~(size_t)255;
    return p;
  };

  __hip_bfloat16* xn    = (__hip_bfloat16*)alloc((size_t)kN * kF * 2); // reused as feat_bf16
  __hip_bfloat16* W1t   = (__hip_bfloat16*)alloc((size_t)kB * kF * 2);
  __hip_bfloat16* W2t   = (__hip_bfloat16*)alloc((size_t)kC * kB * 2);
  float*          norms = (float*)alloc((size_t)kN * 4);
  int*            topk  = (int*)alloc((size_t)kN * kK * 4);
  float*          deg   = (float*)alloc((size_t)kN * 4);
  float*          hW    = (float*)alloc((size_t)kN * kB * 4);
  float*          em1   = (float*)alloc((size_t)kN * kB * 4);
  float*          acc1  = (float*)alloc((size_t)kN * kB * 4);
  float*          hW2   = (float*)alloc((size_t)kN * kC * 4);
  float*          em2   = (float*)alloc((size_t)kN * kC * 4);
  float*          acc2  = (float*)alloc((size_t)kN * kC * 4);

  // 1. normalize + bf16
  hipLaunchKernelGGL(hy_normalize_kernel, dim3(kN), dim3(256), 0, stream, x, xn, norms);

  // 2. transposes
  hipLaunchKernelGGL(hy_transpose_bf16_kernel, dim3((kF * kB) / 256), dim3(256), 0, stream,
                     W1, W1t, kF, kB);
  hipLaunchKernelGGL(hy_transpose_bf16_kernel, dim3((kB * kC) / 256), dim3(256), 0, stream,
                     W2, W2t, kB, kC);

  // 3. fused similarity GEMM + top-8 (never materializes S)
  hipLaunchKernelGGL(hy_s_topk_kernel, dim3(kN / 16), dim3(256), 0, stream, xn, topk);

  // 4. degrees
  hipLaunchKernelGGL(hy_zero_f32_kernel, dim3((kN + 255) / 256), dim3(256), 0, stream,
                     deg, (size_t)kN);
  hipLaunchKernelGGL(hy_deg_kernel, dim3((kN * kK) / 256), dim3(256), 0, stream, topk, deg);

  // 5. hW = diag(norm) * (xn @ W1)   [64 n-blocks of 64 cols, 8 per WG]
  hipLaunchKernelGGL(hy_gemm_bf16_kernel, dim3(kN / 32, (kB / 64) / 8), dim3(256), 0, stream,
                     xn, W1t, hW, kB, kF, kB / 64, norms);

  // 6. layer-1 hypergraph conv
  hipLaunchKernelGGL(hy_edge_mean_kernel, dim3(kN), dim3(256), 0, stream, hW, topk, em1, kB);
  hipLaunchKernelGGL(hy_zero_f32_kernel, dim3(((size_t)kN * kB) / 256), dim3(256), 0, stream,
                     acc1, (size_t)kN * kB);
  hipLaunchKernelGGL(hy_scatter_kernel, dim3(kN), dim3(256), 0, stream, em1, topk, acc1, kB);
  hipLaunchKernelGGL(hy_finish1_kernel, dim3(kN), dim3(256), 0, stream, acc1, deg, b1, xn);

  // 7. layer-2: hW2 = feat @ W2  [single 64-col n-block -> 1 wave per WG]
  hipLaunchKernelGGL(hy_gemm_bf16_kernel, dim3(kN / 32, 1), dim3(32), 0, stream,
                     xn, W2t, hW2, kC, kB, kC / 64, (const float*)nullptr);
  hipLaunchKernelGGL(hy_edge_mean_kernel, dim3(kN), dim3(64), 0, stream, hW2, topk, em2, kC);
  hipLaunchKernelGGL(hy_zero_f32_kernel, dim3(((size_t)kN * kC) / 256), dim3(256), 0, stream,
                     acc2, (size_t)kN * kC);
  hipLaunchKernelGGL(hy_scatter_kernel, dim3(kN), dim3(64), 0, stream, em2, topk, acc2, kC);
  hipLaunchKernelGGL(hy_finish2_kernel, dim3(kN), dim3(64), 0, stream, acc2, deg, b2, out);
}
